// TGAT_26319559590715
// MI455X (gfx1250) — matrix operand order repaired and checked by reading, not compiled, not run on hardware
//
#include <hip/hip_runtime.h>
#include <hip/hip_bf16.h>
#include <cstdint>

// ---------------------------------------------------------------------------
// TGAT forward on gfx1250 (MI455X). All GEMMs via v_wmma_f32_16x16x32_f16.
// Memory-bound (~110 GFLOP vs >=600MB traffic at 23.3TB/s) -> f16 data path,
// f32 accumulation, fused builds/bias/relu/concat, chunked big pass.
// Weight tiles padded to multiples of 64 output cols so the WMMA k-loop is
// completely branch-free (no accumulator shuffling, no SALU ladders).
// ---------------------------------------------------------------------------

typedef __attribute__((ext_vector_type(16))) _Float16 v16h;
typedef __attribute__((ext_vector_type(8)))  _Float16 v8h;
typedef __attribute__((ext_vector_type(8)))  float    v8f;

#define ND    172   // node / edge feature dim
#define TD    100   // time dim
#define QD    272   // qdim = 172+100
#define QDP   288   // qdim padded to mult of 32 (WMMA K dim)
#define NPQ   320   // 272 output cols padded to mult of 64 (20 tiles, 5 groups)
#define KDL   444   // kdim = 172+172+100
#define KDP   448   // kdim padded
#define HD    136   // head dim (272/2)
#define NPE   192   // 172 output cols padded to mult of 64 (12 tiles, 3 groups)
#define EIP   192   // embed as inner dim padded to mult of 32
#define KNB   20    // neighbors
#define CHN   2048  // chunk of n-rows for the big pass

// ---------------- fragment helpers (CDNA5 16-bit WMMA layouts) -------------
__device__ __forceinline__ v16h cat16(v8h lo, v8h hi) {
    v16h v;
#pragma unroll
    for (int i = 0; i < 8; ++i) { v[i] = lo[i]; v[i + 8] = hi[i]; }
    return v;
}

// ---------------- generic WMMA GEMM: C = A(f16) @ Wt^T(f16) + bias ---------
// One wave -> 16x64 output strip. B (weights) always has 4 full 16-col tiles
// per grid.y group (rows of Wt zero-padded), so the k-loop is branch-free.
// A 16x32 f16 frag: lane L: m=L&15; elems i -> k = kt*32 + ((L>>4)<<4) + i
// B 32x16 f16 frag: lane L: n=L&15; elems i -> k = kt*32 + ((L>>4)<<4) + i
//   (Wt stores W^T so each lane reads 16 contiguous halves).
// C/D f32: lane L: n=L&15; VGPR r -> m = r + ((L>>4)<<3).
__global__ __launch_bounds__(256)
void wmma_gemm_kernel(const _Float16* __restrict__ A, int lda,
                      const _Float16* __restrict__ Wt, int ldw,
                      const float* __restrict__ bias,
                      float* __restrict__ Cf, int ldcf,
                      _Float16* __restrict__ Ch, int ldch,
                      int M, int Kp, int NoutStore, int NoutReal, int relu) {
    int lane  = threadIdx.x & 31;
    int wave  = threadIdx.x >> 5;
    int mtile = blockIdx.x * 8 + wave;
    if (mtile * 16 >= M) return;
    int ntg = blockIdx.y * 4;          // 4 full B tiles, always valid (padded)

    const _Float16* pa = A +
        (size_t)(mtile * 16 + (lane & 15)) * lda + ((lane >> 4) << 4);
    const _Float16* pb = Wt +
        (size_t)(ntg * 16 + (lane & 15)) * ldw + ((lane >> 4) << 4);
    const size_t bstep = (size_t)16 * ldw;   // next B tile (16 Wt rows)

    v8f acc0 = {}, acc1 = {}, acc2 = {}, acc3 = {};
    int ktiles = Kp >> 5;
    for (int kt = 0; kt < ktiles; ++kt) {
        v16h a  = cat16(((const v8h*)pa)[0], ((const v8h*)pa)[1]);
        v16h b0 = cat16(((const v8h*)(pb            ))[0], ((const v8h*)(pb            ))[1]);
        v16h b1 = cat16(((const v8h*)(pb + bstep    ))[0], ((const v8h*)(pb + bstep    ))[1]);
        v16h b2 = cat16(((const v8h*)(pb + 2 * bstep))[0], ((const v8h*)(pb + 2 * bstep))[1]);
        v16h b3 = cat16(((const v8h*)(pb + 3 * bstep))[0], ((const v8h*)(pb + 3 * bstep))[1]);
        acc0 = __builtin_amdgcn_wmma_f32_16x16x32_f16(false, a, false, b0, (short)0, acc0, false, false);
        acc1 = __builtin_amdgcn_wmma_f32_16x16x32_f16(false, a, false, b1, (short)0, acc1, false, false);
        acc2 = __builtin_amdgcn_wmma_f32_16x16x32_f16(false, a, false, b2, (short)0, acc2, false, false);
        acc3 = __builtin_amdgcn_wmma_f32_16x16x32_f16(false, a, false, b3, (short)0, acc3, false, false);
        pa += 32;
        pb += 32;
    }

    // store (M is always a multiple of 16 in this pipeline)
    int m0 = mtile * 16 + ((lane >> 4) << 3);
    v8f accs[4] = {acc0, acc1, acc2, acc3};
#pragma unroll
    for (int t = 0; t < 4; ++t) {
        int col = (ntg + t) * 16 + (lane & 15);
        if (col >= NoutStore) continue;
        float bf = (col < NoutReal) ? bias[col] : 0.0f;
#pragma unroll
        for (int r2 = 0; r2 < 8; ++r2) {
            float v = accs[t][r2] + bf;
            if (relu) v = fmaxf(v, 0.0f);
            size_t m = (size_t)(m0 + r2);
            if (Cf) Cf[m * ldcf + col] = v;
            if (Ch) Ch[m * ldch + col] = (_Float16)v;
        }
    }
}

// ---------------- weight packing: W[Din,Dout] f32 -> Wt[Np,Kp] f16 ---------
__global__ void pack_wt_kernel(const float* __restrict__ W, _Float16* __restrict__ Wt,
                               int Din, int Dout, int Kp, int Np) {
    int idx = blockIdx.x * blockDim.x + threadIdx.x;
    int total = Np * Kp;
    if (idx >= total) return;
    int nrow = idx / Kp;
    int k    = idx - nrow * Kp;
    float v = (k < Din && nrow < Dout) ? W[(size_t)k * Dout + nrow] : 0.0f;
    Wt[idx] = (_Float16)v;
}

// ---------------- build K input rows: [node|edge|t2v(dt)|pad] f16 ----------
__global__ void build_k_rows_kernel(_Float16* __restrict__ Xk,
                                    const float* __restrict__ node_src,
                                    const int* __restrict__ node_ids,
                                    const float* __restrict__ edge_feat,
                                    const float* __restrict__ t_center,
                                    const float* __restrict__ t_nbr,
                                    const float* __restrict__ time_w,
                                    const float* __restrict__ time_b,
                                    int M, int K) {
    int r = blockIdx.x;
    if (r >= M) return;
    int n = r / K;
    int nrow = node_ids ? node_ids[r] : r;
    if (nrow < 0) nrow = 0;
    float dt = t_center[n] - t_nbr[r];
    _Float16* out = Xk + (size_t)r * KDP;
    for (int c = threadIdx.x; c < KDP; c += blockDim.x) {
        float v;
        if (c < ND)            v = node_src[(size_t)nrow * ND + c];
        else if (c < 2 * ND)   v = edge_feat[(size_t)r * ND + (c - ND)];
        else if (c < KDL)      { int d = c - 2 * ND; v = __cosf(dt * time_w[d] + time_b[d]); }
        else                   v = 0.0f;
        out[c] = (_Float16)v;
    }
}

// ---------------- build Q input rows: [node|t2v(0)|pad] f16 ----------------
__global__ void build_q_rows_kernel(_Float16* __restrict__ Xq,
                                    const float* __restrict__ node_src,
                                    const int* __restrict__ ids,
                                    const float* __restrict__ time_b,
                                    int N) {
    int n = blockIdx.x;
    if (n >= N) return;
    int nrow = ids ? ids[n] : n;
    if (nrow < 0) nrow = 0;
    _Float16* out = Xq + (size_t)n * QDP;
    for (int c = threadIdx.x; c < QDP; c += blockDim.x) {
        float v;
        if (c < ND)      v = node_src[(size_t)nrow * ND + c];
        else if (c < QD) v = __cosf(time_b[c - ND]);   // t=0 -> cos(b)
        else             v = 0.0f;
        out[c] = (_Float16)v;
    }
}

// ---------------- fill fc1 input tail cols [272..447] with z0 --------------
__global__ void fill_fc1_extra_kernel(_Float16* __restrict__ fc1in,
                                      const float* __restrict__ table,
                                      const int* __restrict__ ids, int N) {
    int n = blockIdx.x;
    if (n >= N) return;
    int nrow = ids[n];
    if (nrow < 0) nrow = 0;
    _Float16* out = fc1in + (size_t)n * KDP + QD;
    for (int c = threadIdx.x; c < (KDP - QD); c += blockDim.x) {
        out[c] = (c < ND) ? (_Float16)table[(size_t)nrow * ND + c] : (_Float16)0.0f;
    }
}

// ---------------- attention core: scores/softmax/weighted-sum, K=20 --------
// grid = N, block = 64 (2 waves = 2 heads). Lane = neighbor slot.
__global__ __launch_bounds__(64)
void attn_kernel(const _Float16* __restrict__ Qh, const _Float16* __restrict__ Kh,
                 const _Float16* __restrict__ Vh, const int* __restrict__ vids,
                 _Float16* __restrict__ outp, int N, int K) {
    int n = blockIdx.x;
    if (n >= N) return;
    int h    = threadIdx.x >> 5;
    int lane = threadIdx.x & 31;
    const float scale = 0.08574929257125442f;  // 1/sqrt(136)

    float s = -1e10f;
    if (lane < K) {
        if (vids[(size_t)n * K + lane] >= 0) {
            const _Float16* q  = Qh + (size_t)n * QD + h * HD;
            const _Float16* kk = Kh + ((size_t)n * K + lane) * QD + h * HD;
            float acc = 0.0f;
            for (int d = 0; d < HD; ++d) acc += (float)q[d] * (float)kk[d];
            s = acc * scale;
        }
    }
    float m = s;
    for (int off = 16; off > 0; off >>= 1) m = fmaxf(m, __shfl_xor(m, off, 32));
    float e = (lane < K) ? __expf(s - m) : 0.0f;
    float tsum = e;
    for (int off = 16; off > 0; off >>= 1) tsum += __shfl_xor(tsum, off, 32);
    float a = e / tsum;

    for (int d = lane; d < HD; d += 32) {
        float acc = 0.0f;
        for (int k = 0; k < K; ++k) {
            float w = __shfl(a, k, 32);
            acc += w * (float)Vh[((size_t)n * K + k) * QD + h * HD + d];
        }
        outp[(size_t)n * QDP + h * HD + d] = (_Float16)acc;
    }
}

// ===========================================================================
extern "C" void kernel_launch(void* const* d_in, const int* in_sizes, int n_in,
                              void* d_out, int out_size, void* d_ws, size_t ws_size,
                              hipStream_t stream) {
    (void)in_sizes; (void)n_in; (void)out_size; (void)ws_size;

    const float* static_feat = (const float*)d_in[0];
    const int*   nids0       = (const int*)d_in[1];
    const int*   nbr_nids0   = (const int*)d_in[2];
    const int*   nbr_nids1   = (const int*)d_in[3];
    const float* nbr_feats0  = (const float*)d_in[4];
    const float* nbr_feats1  = (const float*)d_in[5];
    const float* times0      = (const float*)d_in[6];
    const float* times1      = (const float*)d_in[7];
    const float* nbr_times0  = (const float*)d_in[8];
    const float* nbr_times1  = (const float*)d_in[9];
    const float* time_w      = (const float*)d_in[10];
    const float* time_b      = (const float*)d_in[11];
    const float* Wq          = (const float*)d_in[12];
    const float* bq          = (const float*)d_in[13];
    const float* Wk          = (const float*)d_in[14];
    const float* bk          = (const float*)d_in[15];
    const float* Wv          = (const float*)d_in[16];
    const float* bv          = (const float*)d_in[17];
    const float* Wo          = (const float*)d_in[18];
    const float* bo          = (const float*)d_in[19];
    const float* fc1w        = (const float*)d_in[20];
    const float* fc1b        = (const float*)d_in[21];
    const float* fc2w        = (const float*)d_in[22];
    const float* fc2b        = (const float*)d_in[23];

    const int N0 = 512, N1 = N0 * KNB;           // 512, 10240
    const int MAXN = CHN;                        // max N per invocation
    const int MAXM = CHN * KNB;                  // 40960

    // ---- workspace bump allocator (256B aligned) ----
    char*  base = (char*)d_ws;
    size_t off  = 0;
    auto alloc = [&](size_t bytes) -> void* {
        void* p = base + off;
        off = (off + bytes + 255) & ~(size_t)255;
        return p;
    };

    _Float16 *Wqt[2], *Wkt[2], *Wvt[2], *Wot[2], *F1t[2], *F2t[2];
    for (int l = 0; l < 2; ++l) {
        Wqt[l] = (_Float16*)alloc((size_t)NPQ * QDP * 2);
        Wkt[l] = (_Float16*)alloc((size_t)NPQ * KDP * 2);
        Wvt[l] = (_Float16*)alloc((size_t)NPQ * KDP * 2);
        Wot[l] = (_Float16*)alloc((size_t)NPQ * QDP * 2);
        F1t[l] = (_Float16*)alloc((size_t)NPE * KDP * 2);
        F2t[l] = (_Float16*)alloc((size_t)NPE * EIP * 2);
    }
    float*     z1_0    = (float*)alloc((size_t)N0 * ND * 4);
    float*     z1_1    = (float*)alloc((size_t)N1 * ND * 4);
    _Float16*  Xk      = (_Float16*)alloc((size_t)MAXM * KDP * 2);
    _Float16*  KhB     = (_Float16*)alloc((size_t)MAXM * QD  * 2);
    _Float16*  VhB     = (_Float16*)alloc((size_t)MAXM * QD  * 2);
    _Float16*  Xq      = (_Float16*)alloc((size_t)MAXN * QDP * 2);
    _Float16*  Qh      = (_Float16*)alloc((size_t)MAXN * QD  * 2);
    _Float16*  attnOut = (_Float16*)alloc((size_t)MAXN * QDP * 2);
    _Float16*  fc1in   = (_Float16*)alloc((size_t)MAXN * KDP * 2);
    _Float16*  hbuf    = (_Float16*)alloc((size_t)MAXN * EIP * 2);

    // zero once: attnOut pad cols (272..287) and hbuf pad cols (176..191)
    // are never rewritten by kernels; everything else is fully overwritten.
    hipMemsetAsync(attnOut, 0, (size_t)MAXN * QDP * 2, stream);
    hipMemsetAsync(hbuf,    0, (size_t)MAXN * EIP * 2, stream);

    // ---- pack weights (tiny, ~2.5MB total; L2-resident during GEMMs) ----
    auto pack = [&](const float* W, _Float16* Wt, int Din, int Dout, int Kp, int Np) {
        int total  = Np * Kp;
        int blocks = (total + 255) / 256;
        pack_wt_kernel<<<blocks, 256, 0, stream>>>(W, Wt, Din, Dout, Kp, Np);
    };
    for (int l = 0; l < 2; ++l) {
        pack(Wq   + (size_t)l * QD  * QD, Wqt[l], QD,  QD, QDP, NPQ);
        pack(Wk   + (size_t)l * KDL * QD, Wkt[l], KDL, QD, KDP, NPQ);
        pack(Wv   + (size_t)l * KDL * QD, Wvt[l], KDL, QD, KDP, NPQ);
        pack(Wo   + (size_t)l * QD  * QD, Wot[l], QD,  QD, QDP, NPQ);
        pack(fc1w + (size_t)l * KDL * ND, F1t[l], KDL, ND, KDP, NPE);
        pack(fc2w + (size_t)l * ND  * ND, F2t[l], ND,  ND, EIP, NPE);
    }

    auto gemm = [&](const _Float16* A, int lda, const _Float16* Wt, int ldw,
                    const float* bias, float* Cf, int ldcf, _Float16* Ch, int ldch,
                    int M, int Kp, int Np, int NoutStore, int NoutReal, int relu) {
        dim3 grid((M / 16 + 7) / 8, Np / 64);     // Np padded to mult of 64
        wmma_gemm_kernel<<<grid, 256, 0, stream>>>(A, lda, Wt, ldw, bias,
                                                   Cf, ldcf, Ch, ldch,
                                                   M, Kp, NoutStore, NoutReal, relu);
    };

    // ---- one attention + FFN pass (N query rows) ----
    auto run_pass = [&](int l, int N,
                        const float* qsrc, const int* qids,
                        const float* ksrc, const int* kids,
                        const float* edge, const float* tc, const float* tn,
                        const int* vids, const int* fcids, float* zout) {
        int M = N * KNB;
        build_k_rows_kernel<<<M, 256, 0, stream>>>(Xk, ksrc, kids, edge, tc, tn,
                                                   time_w, time_b, M, KNB);
        build_q_rows_kernel<<<N, 256, 0, stream>>>(Xq, qsrc, qids, time_b, N);
        gemm(Xq, QDP, Wqt[l], QDP, bq + (size_t)l * QD, nullptr, 0, Qh,  QD,
             N, QDP, NPQ, QD, QD, 0);
        gemm(Xk, KDP, Wkt[l], KDP, bk + (size_t)l * QD, nullptr, 0, KhB, QD,
             M, KDP, NPQ, QD, QD, 0);
        gemm(Xk, KDP, Wvt[l], KDP, bv + (size_t)l * QD, nullptr, 0, VhB, QD,
             M, KDP, NPQ, QD, QD, 0);
        attn_kernel<<<N, 64, 0, stream>>>(Qh, KhB, VhB, vids, attnOut, N, KNB);
        // Wo projection -> f16 directly into the first 272 cols of fc1 input
        gemm(attnOut, QDP, Wot[l], QDP, bo + (size_t)l * QD, nullptr, 0, fc1in, KDP,
             N, QDP, NPQ, QD, QD, 0);
        fill_fc1_extra_kernel<<<N, 256, 0, stream>>>(fc1in, static_feat, fcids, N);
        gemm(fc1in, KDP, F1t[l], KDP, fc1b + (size_t)l * ND, nullptr, 0, hbuf, EIP,
             N, KDP, NPE, 176, ND, 1);
        gemm(hbuf, EIP, F2t[l], EIP, fc2b + (size_t)l * ND, zout, ND, nullptr, 0,
             N, EIP, NPE, ND, ND, 0);
    };

    // ---- layer 1, i = 0 : N=512, sources = static features ----
    run_pass(0, N0,
             static_feat, nids0,
             static_feat, nbr_nids0,
             nbr_feats0, times0, nbr_times0,
             nbr_nids0, nids0, z1_0);

    // ---- layer 1, i = 1 : N=10240, chunked by 2048 ----
    for (int c = 0; c < N1 / CHN; ++c) {
        int n0 = c * CHN;
        run_pass(0, CHN,
                 static_feat, nbr_nids0 + n0,                     // z[0][1] = static[nbr_nids0]
                 static_feat, nbr_nids1 + (size_t)n0 * KNB,       // z[0][2]
                 nbr_feats1 + (size_t)n0 * KNB * ND,
                 times1 + n0, nbr_times1 + (size_t)n0 * KNB,
                 nbr_nids1 + (size_t)n0 * KNB,
                 nbr_nids0 + n0,
                 z1_1 + (size_t)n0 * ND);
    }

    // ---- layer 2, i = 0 : N=512, sources = layer-1 outputs (dense f32) ----
    run_pass(1, N0,
             z1_0, nullptr,
             z1_1, nullptr,
             nbr_feats0, times0, nbr_times0,
             nbr_nids0, nids0, (float*)d_out);
}
